// Net_40785009443060
// MI455X (gfx1250) — compile-verified, matrix operand order checked by reading, not run on hardware
//
#include <hip/hip_runtime.h>

#define EMB 32
#define HID 32
#define BATCH 1024
#define SEQ 512

typedef __attribute__((ext_vector_type(16))) _Float16 v16h;
typedef __attribute__((ext_vector_type(8)))  float    v8f;

__device__ __forceinline__ float fast_sigmoid(float x) {
    return 1.0f / (1.0f + __expf(-x));
}
__device__ __forceinline__ float fast_tanh(float x) {
    return 2.0f / (1.0f + __expf(-2.0f * x)) - 1.0f;
}

// Convert 16 f32 from one matrix row into the per-lane element order of a
// CDNA5 16-bit 16x32 A tile (and, symmetrically, a 32x16 B tile):
//   lanes 0-15  hold K = 0..7  and K = 16..23
//   lanes 16-31 hold K = 8..15 and K = 24..31
// caller passes k0 = (lane>>4)*8; elements 0..7 = row[k0..k0+7],
// elements 8..15 = row[k0+16..k0+23].
__device__ __forceinline__ v16h load_tile16(const float* __restrict__ row, int k0) {
    v16h r;
#pragma unroll
    for (int i = 0; i < 8; ++i) {
        r[i]     = (_Float16)row[k0 + i];
        r[i + 8] = (_Float16)row[k0 + 16 + i];
    }
    return r;
}

// One workgroup = one wave32 = one 16-row batch tile carried through all 512
// timesteps (recurrence is private to the tile -> no inter-block sync).
__global__ __launch_bounds__(32)
void lstm_fused_kernel(const int*   __restrict__ x,      // [B, S]
                       const float* __restrict__ emb,    // [VOCAB+1, EMB]
                       const float* __restrict__ w_ih,   // [4H, EMB]
                       const float* __restrict__ w_hh,   // [4H, HID]
                       const float* __restrict__ b_ih,   // [4H]
                       const float* __restrict__ b_hh,   // [4H]
                       const float* __restrict__ fc_w,   // [2, HID]
                       const float* __restrict__ fc_b,   // [2]
                       float*       __restrict__ out) {  // [B, 2]
    __shared__ float hbuf[16 * HID];       // h state, row-major [16][32]

    const int lane = threadIdx.x;
    const int n    = lane & 15;            // column inside a 16-wide tile
    const int k0   = (lane >> 4) * 8;      // K-chunk base for f16 A/B tiles
    const int mb   = (lane >> 4) * 8;      // M base for f32 C tiles
    const int b0   = blockIdx.x * 16;      // batch tile base

    // ---- weights -> f16 WMMA B tiles, resident in VGPRs for all steps ----
    // B tile t covers gate columns [t*16, t*16+16); B[k][nn] = W[t*16+nn][k].
    v16h bIH[8], bHH[8];
#pragma unroll
    for (int t = 0; t < 8; ++t) {
        bIH[t] = load_tile16(&w_ih[(t * 16 + n) * EMB], k0);
        bHH[t] = load_tile16(&w_hh[(t * 16 + n) * HID], k0);
    }
    // gate bias is constant down each C-tile column -> one scalar per lane/tile
    float bias[8];
#pragma unroll
    for (int t = 0; t < 8; ++t)
        bias[t] = b_ih[t * 16 + n] + b_hh[t * 16 + n];

    // ---- zero initial h (LDS) and c (registers) ----
#pragma unroll
    for (int i = 0; i < 16; ++i) hbuf[lane * 16 + i] = 0.0f;
    __syncthreads();

    v8f c0 = {}, c1 = {};
    const int brow = b0 + n;               // batch row this lane gathers for

    for (int s = 0; s < SEQ; ++s) {
        // embedding gather -> A tile (x_e, 16x32)
        const int tok = x[brow * SEQ + s];
        v16h aX = load_tile16(&emb[tok * EMB], k0);
        if (s + 1 < SEQ) {                 // warm L2/L0 for next step's row
            const int tok2 = x[brow * SEQ + s + 1];
            __builtin_prefetch(&emb[tok2 * EMB], 0, 0);
        }
        // previous h -> A tile (16x32), relayout via LDS
        v16h aH = load_tile16(&hbuf[n * HID], k0);
        __syncthreads();                   // reads done before h is rewritten

        // HID=32 -> two 16-wide hidden column groups j; gate tile t = gate*2+j
#pragma unroll
        for (int j = 0; j < 2; ++j) {
            v8f acc[4];
#pragma unroll
            for (int g = 0; g < 4; ++g) {
                const int t = g * 2 + j;
                v8f cseed;
#pragma unroll
                for (int r = 0; r < 8; ++r) cseed[r] = bias[t];
                // gates = bias + x_e*W_ih^T + h*W_hh^T : two chained K=32 WMMAs
                acc[g] = __builtin_amdgcn_wmma_f32_16x16x32_f16(
                    false, aX, false, bIH[t], (short)0, cseed, false, false);
                acc[g] = __builtin_amdgcn_wmma_f32_16x16x32_f16(
                    false, aH, false, bHH[t], (short)0, acc[g], false, false);
            }
            v8f& cj = j ? c1 : c0;
#pragma unroll
            for (int r = 0; r < 8; ++r) {
                const float iv = fast_sigmoid(acc[0][r]);
                const float fv = fast_sigmoid(acc[1][r]);
                const float gv = fast_tanh  (acc[2][r]);
                const float ov = fast_sigmoid(acc[3][r]);
                const float cv = fv * cj[r] + iv * gv;
                cj[r] = cv;
                const float hv = ov * fast_tanh(cv);
                // scatter back into row-major h for next step's A-tile read
                hbuf[(mb + r) * HID + (j * 16 + n)] = hv;
            }
        }
        __syncthreads();                   // h writes visible to next read
    }

    // ---- FC head: out[b, cls] = h_last . fc_w[cls] + fc_b[cls] ----
    {
        const int row = lane & 15;
        const int cls = lane >> 4;         // 2 classes across the half-waves
        float acc = fc_b[cls];
#pragma unroll
        for (int k = 0; k < HID; ++k)
            acc += hbuf[row * HID + k] * fc_w[cls * HID + k];
        out[(b0 + row) * 2 + cls] = acc;
    }
}

extern "C" void kernel_launch(void* const* d_in, const int* in_sizes, int n_in,
                              void* d_out, int out_size, void* d_ws, size_t ws_size,
                              hipStream_t stream) {
    const int*   x    = (const int*)  d_in[0];
    const float* emb  = (const float*)d_in[1];
    const float* w_ih = (const float*)d_in[2];
    const float* w_hh = (const float*)d_in[3];
    const float* b_ih = (const float*)d_in[4];
    const float* b_hh = (const float*)d_in[5];
    const float* fc_w = (const float*)d_in[6];
    const float* fc_b = (const float*)d_in[7];
    float* out = (float*)d_out;
    (void)in_sizes; (void)n_in; (void)out_size; (void)d_ws; (void)ws_size;

    dim3 grid(BATCH / 16);   // 64 independent batch tiles
    dim3 block(32);          // one wave32 per tile
    lstm_fused_kernel<<<grid, block, 0, stream>>>(
        x, emb, w_ih, w_hh, b_ih, b_hh, fc_w, fc_b, out);
}
